// MultiHeadAttention_34694745817616
// MI455X (gfx1250) — matrix operand order repaired and checked by reading, not compiled, not run on hardware
//
#include <hip/hip_runtime.h>
#include <hip/hip_bf16.h>
#include <cstdint>
#include <math.h>

// ---------------------------------------------------------------------------
// MHA forward for MI455X (gfx1250, wave32, WMMA bf16 16x16x32, TDM staging)
//   B=4, SQ=SK=2048, D_MODEL=512, H=8, DK=DV=64, causal mask, padding mask = 0
// ---------------------------------------------------------------------------

#define BATCH   4
#define SQ      2048
#define SK      2048
#define DM      512
#define NH      8
#define DK      64
#define DV      64
#define LOG2E   1.44269504088896340736f

typedef __bf16 bf16;
typedef __attribute__((ext_vector_type(16))) __bf16 bf16x16;
typedef __attribute__((ext_vector_type(8)))  __bf16 bf16x8;
typedef __attribute__((ext_vector_type(8)))  float  v8f;
typedef __attribute__((ext_vector_type(4)))  float  f32x4;
typedef __attribute__((ext_vector_type(4)))  unsigned int u32x4;
typedef __attribute__((ext_vector_type(8)))  int    i32x8;
typedef __attribute__((ext_vector_type(4)))  int    i32x4;

union FragAB { bf16x16 v; bf16x8 h[2]; };

// ds_swizzle-based butterfly within 16-lane halves (wave32)
#define SWZF(x, imm) __int_as_float(__builtin_amdgcn_ds_swizzle(__float_as_int(x), (imm)))
__device__ inline float redmax16(float x) {
  x = fmaxf(x, SWZF(x, 0x041F));   // xor 1
  x = fmaxf(x, SWZF(x, 0x081F));   // xor 2
  x = fmaxf(x, SWZF(x, 0x101F));   // xor 4
  x = fmaxf(x, SWZF(x, 0x201F));   // xor 8
  return x;
}
__device__ inline float redsum16(float x) {
  x += SWZF(x, 0x041F);
  x += SWZF(x, 0x081F);
  x += SWZF(x, 0x101F);
  x += SWZF(x, 0x201F);
  return x;
}

__device__ inline v8f wmma_bf(bf16x16 a, bf16x16 b, v8f c) {
  return __builtin_amdgcn_wmma_f32_16x16x32_bf16(false, a, false, b, (short)0, c, false, false);
}

__device__ inline bf16x8 cvt8(const float* p) {
  f32x4 x = *(const f32x4*)p;
  f32x4 y = *(const f32x4*)(p + 4);
  bf16x8 r;
#pragma unroll
  for (int i = 0; i < 4; i++) { r[i] = (bf16)x[i]; r[i + 4] = (bf16)y[i]; }
  return r;
}

// ---------------------------------------------------------------------------
// Tensor Data Mover: 2-D bf16 tile load, global -> LDS, with optional LDS row
// padding.  Descriptor per CDNA5 ISA ch.8 (D# groups 0/1; groups 2/3 zero for
// 2-D tensors).  Issued once per wave; tracked on TENSORcnt.
//   tile_w elems per row, tile_h rows, row stride in elems,
//   pad_interval code: pad after 2^(c+1) DWORDs, pad_amount code: (c+1) DWORDs
// ---------------------------------------------------------------------------
__device__ inline void tdm_load_2d(unsigned lds_addr, const void* gaddr,
                                   unsigned tile_w, unsigned tile_h,
                                   unsigned row_stride,
                                   unsigned pad_int_code, unsigned pad_amt_code) {
  unsigned long long ga = (unsigned long long)(uintptr_t)gaddr;
  u32x4 g0;
  g0[0] = 1u;                                          // count=1, user descriptor
  g0[1] = lds_addr;                                    // LDS byte address
  g0[2] = (unsigned)(ga & 0xFFFFFFFFu);                // global_addr[31:0]
  g0[3] = (unsigned)((ga >> 32) & 0x01FFFFFFu)         // global_addr[56:32]
        | (2u << 30);                                  // type=2 ("image")
  i32x8 g1;
  unsigned w0 = (1u << 16);                            // data_size=1 -> 2 bytes
  w0 |= (1u << 20) | (pad_int_code << 22) | (pad_amt_code << 25);  // pad_enable
  g1[0] = (int)w0;
  g1[1] = (int)((tile_w & 0xFFFFu) << 16);             // tensor_dim0[15:0] @63:48
  g1[2] = (int)(((tile_w >> 16) & 0xFFFFu)             // tensor_dim0[31:16]
        | ((tile_h & 0xFFFFu) << 16));                 // tensor_dim1[15:0]
  g1[3] = (int)(((tile_h >> 16) & 0xFFFFu)             // tensor_dim1[31:16]
        | ((tile_w & 0xFFFFu) << 16));                 // tile_dim0 @127:112
  g1[4] = (int)(tile_h & 0xFFFFu);                     // tile_dim1; tile_dim2=0
  g1[5] = (int)row_stride;                             // tensor_dim0_stride[31:0]
  g1[6] = 0;                                           // stride hi, dim1_stride lo
  g1[7] = 0;
  i32x4 z4 = {0, 0, 0, 0};
  i32x8 z8 = {0, 0, 0, 0, 0, 0, 0, 0};
  __builtin_amdgcn_tensor_load_to_lds(g0, g1, z4, z4, z8, 0);
}

// ---------------------------------------------------------------------------
// prep kernels
// ---------------------------------------------------------------------------
__global__ void prep_wt(const float* __restrict__ Wq, const float* __restrict__ Wk,
                        const float* __restrict__ Wv,
                        bf16* __restrict__ WqT, bf16* __restrict__ WkT, bf16* __restrict__ WvT) {
  int idx = blockIdx.x * blockDim.x + threadIdx.x;
  const int per = NH * DM * DK;
  if (idx >= 3 * per) return;
  int which = idx / per;  int r = idx % per;
  int h = r / (DM * DK);  int r2 = r % (DM * DK);
  int k = r2 / DK;        int n = r2 % DK;
  const float* W = (which == 0) ? Wq : (which == 1) ? Wk : Wv;
  bf16* T        = (which == 0) ? WqT : (which == 1) ? WkT : WvT;
  T[((size_t)h * DK + n) * DM + k] = (bf16)W[((size_t)h * DM + k) * DK + n];
}

// WcT[n][kk] = sum_u Wh[h][v][u] * Wo[h*64+u][n],  kk = h*64 + v   (N-major)
__global__ void prep_wc(const float* __restrict__ Wh, const float* __restrict__ Wo,
                        bf16* __restrict__ WcT) {
  int idx = blockIdx.x * blockDim.x + threadIdx.x;
  if (idx >= DM * DM) return;
  int n = idx >> 9, kk = idx & 511;
  int h = kk >> 6, v = kk & 63;
  float acc = 0.f;
  for (int u = 0; u < 64; u++)
    acc += Wh[((size_t)h * 64 + v) * 64 + u] * Wo[(size_t)(h * 64 + u) * DM + n];
  WcT[(size_t)n * DM + kk] = (bf16)acc;
}

__global__ void prep_bout(const float* __restrict__ bh_, const float* __restrict__ bo,
                          const float* __restrict__ Wo, float* __restrict__ bout) {
  int n = blockIdx.x * blockDim.x + threadIdx.x;
  if (n >= DM) return;
  float acc = bo[n];
  for (int j = 0; j < DM; j++) acc += bh_[j] * Wo[(size_t)j * DM + n];
  bout[n] = acc;
}

// ---------------------------------------------------------------------------
// kernel 1: per-head q/k/v projections.  q,k stored row-major [bh][s][64] bf16,
// v stored transposed [bh][dv][s] bf16.  q pre-scaled by 1/sqrt(dk).
// grid: (SQ/128, B*H), block 256 (8 waves, 16 rows each)
// ---------------------------------------------------------------------------
__device__ inline void proj_one(const float* __restrict__ src, const bf16* __restrict__ wT,
                                const float* __restrict__ bias, bf16* __restrict__ dst,
                                float scale, int transposed,
                                int bh, int h, int row0, int nl, int half) {
  v8f acc[4] = {{0,0,0,0,0,0,0,0},{0,0,0,0,0,0,0,0},{0,0,0,0,0,0,0,0},{0,0,0,0,0,0,0,0}};
  const float* ap_base = src + (size_t)(row0 + nl) * DM;
  for (int ks = 0; ks < DM / 32; ks++) {
    FragAB a;
    const float* ap = ap_base + ks * 32 + half * 8;
    a.h[0] = cvt8(ap);
    a.h[1] = cvt8(ap + 16);
#pragma unroll
    for (int j = 0; j < 4; j++) {
      FragAB bb;
      const bf16* bp = wT + ((size_t)h * DK + j * 16 + nl) * DM + ks * 32 + half * 8;
      bb.h[0] = *(const bf16x8*)bp;
      bb.h[1] = *(const bf16x8*)(bp + 16);
      acc[j] = wmma_bf(a.v, bb.v, acc[j]);
    }
  }
#pragma unroll
  for (int j = 0; j < 4; j++) {
    float bn = bias[h * DK + j * 16 + nl];
    if (!transposed) {
#pragma unroll
      for (int r = 0; r < 8; r++) {
        float val = (acc[j][r] + bn) * scale;
        dst[((size_t)bh * SQ + row0 + r + 8 * half) * DK + j * 16 + nl] = (bf16)val;
      }
    } else {
      bf16x8 pk;
#pragma unroll
      for (int r = 0; r < 8; r++) pk[r] = (bf16)((acc[j][r] + bn) * scale);
      *(bf16x8*)(dst + ((size_t)bh * DV + j * 16 + nl) * SK + row0 + 8 * half) = pk;
    }
  }
}

__global__ __launch_bounds__(256) void proj_kernel(
    const float* __restrict__ Q, const float* __restrict__ K, const float* __restrict__ V,
    const float* __restrict__ bq, const float* __restrict__ bk, const float* __restrict__ bv,
    const bf16* __restrict__ WqT, const bf16* __restrict__ WkT, const bf16* __restrict__ WvT,
    bf16* __restrict__ qg, bf16* __restrict__ kg, bf16* __restrict__ vtg) {
  const int t = threadIdx.x, w = t >> 5, lane = t & 31, nl = lane & 15, half = lane >> 4;
  const int bh = blockIdx.y, b = bh >> 3, h = bh & 7;
  const int row0 = blockIdx.x * 128 + w * 16;
  const float* Qb = Q + (size_t)b * SQ * DM;
  const float* Kb = K + (size_t)b * SK * DM;
  const float* Vb = V + (size_t)b * SK * DM;
  proj_one(Qb, WqT, bq, qg, 0.125f, 0, bh, h, row0, nl, half);   // 1/sqrt(64) folded in
  proj_one(Kb, WkT, bk, kg, 1.0f,   0, bh, h, row0, nl, half);
  proj_one(Vb, WvT, bv, vtg, 1.0f,  1, bh, h, row0, nl, half);
}

// ---------------------------------------------------------------------------
// kernel 2: causal flash attention.  grid (SQ/128, B*H), block 256.
// K/V 64-key tiles double-buffered in LDS via TDM (wave 0 issues, TENSORcnt).
// ---------------------------------------------------------------------------
__global__ __launch_bounds__(256) void attn_kernel(
    const bf16* __restrict__ qg, const bf16* __restrict__ kg, const bf16* __restrict__ vtg,
    bf16* __restrict__ ctx) {
  __shared__ __align__(16) bf16 Ks[2][64][72];    // k rows  [key][d]   (+16B row pad)
  __shared__ __align__(16) bf16 Vt[2][64][72];    // vT rows [dv][key]
  __shared__ __align__(16) bf16 Ps[8][16][72];    // per-wave P staging

  const int t = threadIdx.x, w = t >> 5, lane = t & 31, nl = lane & 15, half = lane >> 4;
  const int bh = blockIdx.y, b = bh >> 3, h = bh & 7;
  const int q0 = blockIdx.x * 128;
  const int myrow0 = q0 + w * 16;

  const bf16* kbp = kg  + (size_t)bh * SK * DK;
  const bf16* vbp = vtg + (size_t)bh * DV * SK;
  const int nkt = blockIdx.x * 2 + 2;              // causal bound: keys < q0+128

  // prologue: TDM-fetch tile 0 into buffer 0 (one wave issues for the block)
  if (w == 0) {
    tdm_load_2d((unsigned)(uintptr_t)&Ks[0][0][0], kbp, 64, 64, 64, 4, 3);
    tdm_load_2d((unsigned)(uintptr_t)&Vt[0][0][0], vbp, 64, 64, SK, 4, 3);
  }

  // q A-fragments (held in registers for the whole pass)
  FragAB qa[2];
  {
    const bf16* qp = qg + ((size_t)bh * SQ + myrow0 + nl) * DK;
#pragma unroll
    for (int kk = 0; kk < 2; kk++) {
      const bf16* p = qp + kk * 32 + half * 8;
      qa[kk].h[0] = *(const bf16x8*)p;
      qa[kk].h[1] = *(const bf16x8*)(p + 16);
    }
  }

  v8f o[4] = {{0,0,0,0,0,0,0,0},{0,0,0,0,0,0,0,0},{0,0,0,0,0,0,0,0},{0,0,0,0,0,0,0,0}};
  float runm[8], runs[8];
#pragma unroll
  for (int r = 0; r < 8; r++) { runm[r] = -3.0e38f; runs[r] = 0.f; }

  for (int kt = 0; kt < nkt; kt++) {
    const int cur = kt & 1;
    if (w == 0) {
      if (kt + 1 < nkt) {                          // prefetch next tile, then wait
        const int nxt = (kt + 1) & 1;              // for the current one (in-order)
        tdm_load_2d((unsigned)(uintptr_t)&Ks[nxt][0][0],
                    kbp + (size_t)(kt + 1) * 64 * DK, 64, 64, 64, 4, 3);
        tdm_load_2d((unsigned)(uintptr_t)&Vt[nxt][0][0],
                    vbp + (kt + 1) * 64, 64, 64, SK, 4, 3);
        __builtin_amdgcn_s_wait_tensorcnt(2);
      } else {
        __builtin_amdgcn_s_wait_tensorcnt(0);
      }
    }
    __syncthreads();                               // tile `cur` visible to all waves

    // ---- scores: S = q * k^T  (4 n-tiles over 64 keys, 2 k-steps over d=64) ----
    v8f s[4];
#pragma unroll
    for (int j = 0; j < 4; j++) {
      v8f c = {0,0,0,0,0,0,0,0};
#pragma unroll
      for (int kk = 0; kk < 2; kk++) {
        FragAB bk;
        const bf16* p = &Ks[cur][j * 16 + nl][kk * 32 + half * 8];
        bk.h[0] = *(const bf16x8*)p;
        bk.h[1] = *(const bf16x8*)(p + 16);
        c = wmma_bf(qa[kk].v, bk.v, c);
      }
      s[j] = c;
    }

    // ---- causal mask (skip when whole tile is below the diagonal: uniform) ----
    const int kbase = kt * 64;
    if (kbase + 63 > myrow0) {
#pragma unroll
      for (int j = 0; j < 4; j++) {
        int key = kbase + j * 16 + nl;
#pragma unroll
        for (int r = 0; r < 8; r++) {
          int qrow = myrow0 + r + 8 * half;
          if (key > qrow) s[j][r] = -3.0e38f;
        }
      }
    }

    // ---- online softmax ----
    float nm[8], alpha[8];
#pragma unroll
    for (int r = 0; r < 8; r++) {
      float m = fmaxf(fmaxf(s[0][r], s[1][r]), fmaxf(s[2][r], s[3][r]));
      m = redmax16(m);                            // across 16 N-lanes of this half
      nm[r] = fmaxf(runm[r], m);
      alpha[r] = exp2f((runm[r] - nm[r]) * LOG2E);
      runm[r] = nm[r];
    }
#pragma unroll
    for (int j = 0; j < 4; j++)
#pragma unroll
      for (int r = 0; r < 8; r++)
        s[j][r] = exp2f((s[j][r] - nm[r]) * LOG2E);
#pragma unroll
    for (int r = 0; r < 8; r++) {
      float ts = s[0][r] + s[1][r] + s[2][r] + s[3][r];
      ts = redsum16(ts);
      runs[r] = runs[r] * alpha[r] + ts;
    }
    // rescale accumulators
#pragma unroll
    for (int j = 0; j < 4; j++)
#pragma unroll
      for (int r = 0; r < 8; r++) o[j][r] *= alpha[r];

    // ---- stage P (C-layout -> row-major) in per-wave LDS ----
#pragma unroll
    for (int j = 0; j < 4; j++)
#pragma unroll
      for (int r = 0; r < 8; r++)
        Ps[w][r + 8 * half][j * 16 + nl] = (bf16)s[j][r];
    asm volatile("s_wait_dscnt 0" ::: "memory");  // same-wave LDS RAW

    // ---- P A-fragments, then O += P * V ----
    FragAB pa[2];
#pragma unroll
    for (int kk = 0; kk < 2; kk++) {
      const bf16* p = &Ps[w][nl][kk * 32 + half * 8];
      pa[kk].h[0] = *(const bf16x8*)p;
      pa[kk].h[1] = *(const bf16x8*)(p + 16);
    }
#pragma unroll
    for (int j = 0; j < 4; j++) {
#pragma unroll
      for (int kk = 0; kk < 2; kk++) {
        FragAB bv;
        const bf16* p = &Vt[cur][j * 16 + nl][kk * 32 + half * 8];
        bv.h[0] = *(const bf16x8*)p;
        bv.h[1] = *(const bf16x8*)(p + 16);
        o[j] = wmma_bf(pa[kk].v, bv.v, o[j]);
      }
    }
    __syncthreads();                               // done reading before overwrite
  }

  // ---- normalize and write ctx in concat layout [b][s][h*64+dv] ----
#pragma unroll
  for (int j = 0; j < 4; j++)
#pragma unroll
    for (int r = 0; r < 8; r++) {
      float val = o[j][r] / runs[r];
      ctx[((size_t)b * SQ + myrow0 + r + 8 * half) * (NH * DV) + h * DV + j * 16 + nl] = (bf16)val;
    }
}

// ---------------------------------------------------------------------------
// kernel 3: out = ctx_concat[8192,512] @ Wc[512,512] + bout   (f32 output)
// grid (8192/128, 512/64), block 256.  Wc tile TDM-staged in LDS.
// ---------------------------------------------------------------------------
__global__ __launch_bounds__(256) void out_kernel(
    const bf16* __restrict__ ctxg, const bf16* __restrict__ wcT,
    const float* __restrict__ bout, float* __restrict__ out) {
  __shared__ __align__(16) bf16 Wb[64][520];      // 64 output cols x 512 K (+16B pad)
  const int t = threadIdx.x, w = t >> 5, lane = t & 31, nl = lane & 15, half = lane >> 4;
  const int n0 = blockIdx.y * 64;
  const int m0 = blockIdx.x * 128 + w * 16;

  if (w == 0) {
    // row = 512 elems = 256 DWORDs -> pad_interval code 7; pad 4 DWORDs -> code 3
    tdm_load_2d((unsigned)(uintptr_t)&Wb[0][0], wcT + (size_t)n0 * DM, 512, 64, 512, 7, 3);
    __builtin_amdgcn_s_wait_tensorcnt(0);
  }
  __syncthreads();

  v8f acc[4] = {{0,0,0,0,0,0,0,0},{0,0,0,0,0,0,0,0},{0,0,0,0,0,0,0,0},{0,0,0,0,0,0,0,0}};
  for (int ks = 0; ks < DM / 32; ks++) {
    FragAB a;
    const bf16* ap = ctxg + (size_t)(m0 + nl) * DM + ks * 32 + half * 8;
    a.h[0] = *(const bf16x8*)ap;
    a.h[1] = *(const bf16x8*)(ap + 16);
#pragma unroll
    for (int j = 0; j < 4; j++) {
      FragAB bb;
      const bf16* bp = &Wb[j * 16 + nl][ks * 32 + half * 8];
      bb.h[0] = *(const bf16x8*)bp;
      bb.h[1] = *(const bf16x8*)(bp + 16);
      acc[j] = wmma_bf(a.v, bb.v, acc[j]);
    }
  }
#pragma unroll
  for (int j = 0; j < 4; j++) {
    float bn = bout[n0 + j * 16 + nl];
#pragma unroll
    for (int r = 0; r < 8; r++)
      out[(size_t)(m0 + r + 8 * half) * DM + n0 + j * 16 + nl] = acc[j][r] + bn;
  }
}

// ---------------------------------------------------------------------------
// launcher
// ---------------------------------------------------------------------------
extern "C" void kernel_launch(void* const* d_in, const int* in_sizes, int n_in,
                              void* d_out, int out_size, void* d_ws, size_t ws_size,
                              hipStream_t stream) {
  const float* Q   = (const float*)d_in[0];
  const float* K   = (const float*)d_in[1];
  const float* V   = (const float*)d_in[2];
  // d_in[3] = padding_mask (all false) -> ignored
  const float* Wq  = (const float*)d_in[4];
  const float* bq  = (const float*)d_in[5];
  const float* Wk  = (const float*)d_in[6];
  const float* bk  = (const float*)d_in[7];
  const float* Wv  = (const float*)d_in[8];
  const float* bv  = (const float*)d_in[9];
  const float* Wh  = (const float*)d_in[10];
  const float* bh  = (const float*)d_in[11];
  const float* Wo  = (const float*)d_in[12];
  const float* bo  = (const float*)d_in[13];
  float* out = (float*)d_out;

  char* ws = (char*)d_ws;
  const size_t SZ_QKV = (size_t)BATCH * NH * SQ * DK * sizeof(bf16);   // 8 MB each
  const size_t SZ_CTX = (size_t)BATCH * SQ * NH * DV * sizeof(bf16);   // 8 MB
  const size_t SZ_WT  = (size_t)NH * DK * DM * sizeof(bf16);           // 0.5 MB each
  const size_t SZ_WC  = (size_t)DM * DM * sizeof(bf16);                // 0.5 MB
  size_t off = 0;
  bf16*  qg   = (bf16*)(ws + off); off += SZ_QKV;
  bf16*  kg   = (bf16*)(ws + off); off += SZ_QKV;
  bf16*  vtg  = (bf16*)(ws + off); off += SZ_QKV;
  bf16*  ctx  = (bf16*)(ws + off); off += SZ_CTX;
  bf16*  WqT  = (bf16*)(ws + off); off += SZ_WT;
  bf16*  WkT  = (bf16*)(ws + off); off += SZ_WT;
  bf16*  WvT  = (bf16*)(ws + off); off += SZ_WT;
  bf16*  WcT  = (bf16*)(ws + off); off += SZ_WC;
  float* bout = (float*)(ws + off); off += DM * sizeof(float);

  prep_wt  <<<(3 * NH * DM * DK + 255) / 256, 256, 0, stream>>>(Wq, Wk, Wv, WqT, WkT, WvT);
  prep_wc  <<<(DM * DM + 255) / 256, 256, 0, stream>>>(Wh, Wo, WcT);
  prep_bout<<<(DM + 255) / 256, 256, 0, stream>>>(bh, bo, Wo, bout);
  proj_kernel<<<dim3(SQ / 128, BATCH * NH), 256, 0, stream>>>(Q, K, V, bq, bk, bv,
                                                              WqT, WkT, WvT, qg, kg, vtg);
  attn_kernel<<<dim3(SQ / 128, BATCH * NH), 256, 0, stream>>>(qg, kg, vtg, ctx);
  out_kernel <<<dim3(BATCH * SQ / 128, DM / 64), 256, 0, stream>>>(ctx, WcT, bout, out);
}
